// LocalAggregation_84052509982736
// MI455X (gfx1250) — matrix operand order repaired and checked by reading, not compiled, not run on hardware
//
#include <hip/hip_runtime.h>
#include <hip/hip_fp16.h>

typedef __attribute__((ext_vector_type(16))) _Float16 v16h;
typedef __attribute__((ext_vector_type(8)))  _Float16 v8h;
typedef __attribute__((ext_vector_type(8)))  float    v8f;
typedef __attribute__((ext_vector_type(4)))  int      v4i;

#define N_PTS 65536
#define C_IN  64
#define K_NB  32
#define KP1   96    // layer-1 input dim padded 67 -> 96 (3 chunks of 32)
#define H1    128
#define H2    256
#define BN_EPS 1e-5f

// ---------------------------------------------------------------------------
// CDNA5 async global->LDS path (ASYNCcnt-tracked), with safe fallback.
// Probe result (round 2): param 1 of the b128 builtin is `v4i*` (generic AS).
#if defined(__has_builtin)
#  if __has_builtin(__builtin_amdgcn_global_load_async_to_lds_b128) && \
      __has_builtin(__builtin_amdgcn_s_wait_asynccnt)
#    define USE_ASYNC_LDS 1
#  endif
#endif
#ifndef USE_ASYNC_LDS
#  define USE_ASYNC_LDS 0
#endif

__device__ __forceinline__ void copy16_g2l(void* lds_dst, const void* gsrc) {
#if USE_ASYNC_LDS
    __builtin_amdgcn_global_load_async_to_lds_b128(
        (v4i*)gsrc, (v4i*)lds_dst, /*imm offset*/ 0, /*cpol*/ 0);
#else
    *(v8h*)lds_dst = *(const v8h*)gsrc;
#endif
}

__device__ __forceinline__ void staging_fence() {
#if USE_ASYNC_LDS
    __builtin_amdgcn_s_wait_asynccnt(0);
#endif
    __syncthreads();
}

// ---------------------------------------------------------------------------
// A-fragment (16-bit A 16x32): lanes 0-15 K={0..15}
// (kbase=0), lanes 16-31 K={16..31} (kbase=16). base already includes
// row*lda + kc*32 + kbase; reads land on ds_load_b128 when base is LDS.
__device__ __forceinline__ v16h load_a_frag(const _Float16* base) {
    v8h lo = *(const v8h*)(base);
    v8h hi = *(const v8h*)(base + 8);
    v16h a;
#pragma unroll
    for (int i = 0; i < 8; ++i) { a[i] = lo[i]; a[i + 8] = hi[i]; }
    return a;
}

// ---------------------------------------------------------------------------
// Prep: W1 [128,67]->f16 [128,96] zero-padded, W2 [256,128]->f16, and zero
// the BN-stat accumulators (re-zeroed every launch: graph-replay safe).
__global__ void prep_kernel(const float* __restrict__ W1,
                            const float* __restrict__ W2,
                            _Float16* __restrict__ W1h,
                            _Float16* __restrict__ W2h,
                            float* __restrict__ stats) {
    const int T1 = H1 * KP1;
    const int T2 = H2 * H1;
    const int T3 = 2 * (H1 + H2);
    const int total = T1 + T2 + T3;
    for (int i = blockIdx.x * blockDim.x + threadIdx.x; i < total;
         i += gridDim.x * blockDim.x) {
        if (i < T1) {
            int r = i / KP1, c = i % KP1;
            W1h[i] = (c < C_IN + 3) ? (_Float16)W1[r * (C_IN + 3) + c]
                                    : (_Float16)0.f;
        } else if (i < T1 + T2) {
            int j = i - T1;
            W2h[j] = (_Float16)W2[j];
        } else {
            stats[i - T1 - T2] = 0.f;
        }
    }
}

// ---------------------------------------------------------------------------
// Gather + neighbor max-pool. One wave per point (wave32).
//  - rel-xyz: one neighbor per lane (3 loads/lane), 5-step __shfl_xor max tree.
//  - features: 2 neighbors/iter; each half-wave loads one neighbor's 64-float
//    row as float4 (b128, fully coalesced, L2-resident: x is 16MB << 192MB L2),
//    final cross-half combine with __shfl_xor(.,16).
// Output: f16 [N,96], channels 67..95 zero-padded.
__global__ void gather_pool_kernel(const float* __restrict__ p,
                                   const float* __restrict__ x,
                                   const int* __restrict__ idx,
                                   _Float16* __restrict__ pooledh) {
    const int lane = threadIdx.x & 31;
    const int wv   = threadIdx.x >> 5;
    const int n    = blockIdx.x * 8 + wv;

    const int myidx = idx[n * K_NB + lane];   // neighbor owned by this lane

    // --- rel xyz max (lane-parallel over neighbors) ---
    const float pnx = p[n * 3 + 0];           // uniform per wave -> scalarized
    const float pny = p[n * 3 + 1];
    const float pnz = p[n * 3 + 2];
    float rx = p[myidx * 3 + 0] - pnx;
    float ry = p[myidx * 3 + 1] - pny;
    float rz = p[myidx * 3 + 2] - pnz;
#pragma unroll
    for (int off = 16; off >= 1; off >>= 1) {
        rx = fmaxf(rx, __shfl_xor(rx, off, 32));
        ry = fmaxf(ry, __shfl_xor(ry, off, 32));
        rz = fmaxf(rz, __shfl_xor(rz, off, 32));
    }

    // --- feature max: 2 neighbors per iteration, float4 per lane ---
    const int sub = lane & 15;        // float4 chunk within a 64-float row
    const int grp = lane >> 4;        // which of the two neighbors this half does
    float4 xm = make_float4(-3.4e38f, -3.4e38f, -3.4e38f, -3.4e38f);
#pragma unroll 4
    for (int k = 0; k < K_NB; k += 2) {
        const int j = __shfl(myidx, k + grp, 32);
        const float4 v = *(const float4*)(x + j * C_IN + sub * 4);
        xm.x = fmaxf(xm.x, v.x);
        xm.y = fmaxf(xm.y, v.y);
        xm.z = fmaxf(xm.z, v.z);
        xm.w = fmaxf(xm.w, v.w);
    }
    xm.x = fmaxf(xm.x, __shfl_xor(xm.x, 16, 32));
    xm.y = fmaxf(xm.y, __shfl_xor(xm.y, 16, 32));
    xm.z = fmaxf(xm.z, __shfl_xor(xm.z, 16, 32));
    xm.w = fmaxf(xm.w, __shfl_xor(xm.w, 16, 32));

    _Float16* row = pooledh + n * KP1;
    if (lane == 0) {
        row[0] = (_Float16)rx;
        row[1] = (_Float16)ry;
        row[2] = (_Float16)rz;
    }
    if (grp == 0) {                            // channels 3 + 4*sub .. +3
        row[3 + sub * 4 + 0] = (_Float16)xm.x;
        row[3 + sub * 4 + 1] = (_Float16)xm.y;
        row[3 + sub * 4 + 2] = (_Float16)xm.z;
        row[3 + sub * 4 + 3] = (_Float16)xm.w;
    } else {                                   // zero-pad channels 67..95
        row[67 + sub] = (_Float16)0.f;
        if (sub < 13) row[83 + sub] = (_Float16)0.f;
    }
}

// ---------------------------------------------------------------------------
// GEMM1: h1 = pooled[N,96] @ W1h[128,96]^T -> f32 [N,128] with fused BN-stat
// accumulation. Block = 256 thr = 8 waves covering 128 rows. The A panel
// (128x96 f16 = 24KB) is staged into LDS ONCE per block (async-to-LDS), then
// all 8 waves (one 16-col N-tile each) read fragments via ds_load_b128.
__global__ void gemm1_kernel(const _Float16* __restrict__ Ah,
                             const _Float16* __restrict__ Bh,
                             float* __restrict__ Cout,
                             float* __restrict__ csum,
                             float* __restrict__ csq) {
    __shared__ _Float16 As[128 * KP1];        // 24 KB
    const int tid   = threadIdx.x;
    const int lane  = tid & 31;
    const int wv    = tid >> 5;               // N-tile
    const int col16 = lane & 15;
    const int half  = (lane >> 4) & 1;
    const int mBase = blockIdx.x * 128;

    // Stage A panel: 1536 x 16B chunks, 6 per thread.
    {
        const _Float16* gsrc = Ah + (size_t)mBase * KP1;
#pragma unroll
        for (int c = 0; c < 6; ++c) {
            const int e = (tid + c * 256) * 8;
            copy16_g2l(&As[e], gsrc + e);
        }
    }

    // B fragment: 16 contiguous halves along K per lane (same K order as the A fragment).
    v16h bfrag[3];
    {
        const _Float16* brow = Bh + (wv * 16 + col16) * KP1 + (half ? 16 : 0);
#pragma unroll
        for (int kc = 0; kc < 3; ++kc)
            bfrag[kc] = *(const v16h*)(brow + kc * 32);
    }

    staging_fence();

    float s = 0.f, sq = 0.f;
#pragma unroll
    for (int mt = 0; mt < 8; ++mt) {
        const _Float16* arow = &As[(mt * 16 + col16) * KP1 + (half ? 16 : 0)];
        v8f acc = {};
#pragma unroll
        for (int kc = 0; kc < 3; ++kc) {
            v16h a = load_a_frag(arow + kc * 32);
            acc = __builtin_amdgcn_wmma_f32_16x16x32_f16(
                false, a, false, bfrag[kc], (short)0, acc, false, false);
        }
        float* crow = Cout + (mBase + mt * 16 + half * 8) * H1 + wv * 16 + col16;
#pragma unroll
        for (int r = 0; r < 8; ++r) {
            float v = acc[r];
            crow[r * H1] = v;
            s += v; sq += v * v;
        }
    }
    s  += __shfl_xor(s, 16, 32);
    sq += __shfl_xor(sq, 16, 32);
    if (!half) {
        atomicAdd(&csum[wv * 16 + col16], s);
        atomicAdd(&csq [wv * 16 + col16], sq);
    }
}

// ---------------------------------------------------------------------------
// GEMM2: h2 = h1f16[N,128] @ W2h[256,128]^T -> f32 [N,256] (into d_out) with
// fused BN stats. 8 waves x 2 N-tiles each; A panel (128x128 f16 = 32KB)
// staged once per block into LDS; A fragments reused across both N-tiles.
__global__ void gemm2_kernel(const _Float16* __restrict__ Ah,
                             const _Float16* __restrict__ Bh,
                             float* __restrict__ Cout,
                             float* __restrict__ csum,
                             float* __restrict__ csq) {
    __shared__ _Float16 As[128 * H1];         // 32 KB
    const int tid   = threadIdx.x;
    const int lane  = tid & 31;
    const int wv    = tid >> 5;
    const int col16 = lane & 15;
    const int half  = (lane >> 4) & 1;
    const int mBase = blockIdx.x * 128;

    // Stage A panel: 2048 x 16B chunks, 8 per thread.
    {
        const _Float16* gsrc = Ah + (size_t)mBase * H1;
#pragma unroll
        for (int c = 0; c < 8; ++c) {
            const int e = (tid + c * 256) * 8;
            copy16_g2l(&As[e], gsrc + e);
        }
    }

    v16h bfrag[2][4];
#pragma unroll
    for (int t = 0; t < 2; ++t) {
        const _Float16* brow =
            Bh + ((wv * 2 + t) * 16 + col16) * H1 + (half ? 16 : 0);
#pragma unroll
        for (int kc = 0; kc < 4; ++kc)
            bfrag[t][kc] = *(const v16h*)(brow + kc * 32);
    }

    staging_fence();

    float s[2] = {0.f, 0.f}, sq[2] = {0.f, 0.f};
    for (int mt = 0; mt < 8; ++mt) {
        const _Float16* arow = &As[(mt * 16 + col16) * H1 + (half ? 16 : 0)];
        v16h a[4];
#pragma unroll
        for (int kc = 0; kc < 4; ++kc) a[kc] = load_a_frag(arow + kc * 32);
#pragma unroll
        for (int t = 0; t < 2; ++t) {
            v8f acc = {};
#pragma unroll
            for (int kc = 0; kc < 4; ++kc)
                acc = __builtin_amdgcn_wmma_f32_16x16x32_f16(
                    false, a[kc], false, bfrag[t][kc], (short)0, acc, false,
                    false);
            float* crow = Cout + (mBase + mt * 16 + half * 8) * H2 +
                          (wv * 2 + t) * 16 + col16;
#pragma unroll
            for (int r = 0; r < 8; ++r) {
                float v = acc[r];
                crow[r * H2] = v;
                s[t] += v; sq[t] += v * v;
            }
        }
    }
#pragma unroll
    for (int t = 0; t < 2; ++t) {
        float ss = s[t]  + __shfl_xor(s[t], 16, 32);
        float qq = sq[t] + __shfl_xor(sq[t], 16, 32);
        if (!half) {
            atomicAdd(&csum[(wv * 2 + t) * 16 + col16], ss);
            atomicAdd(&csq [(wv * 2 + t) * 16 + col16], qq);
        }
    }
}

// ---------------------------------------------------------------------------
// BN (training batch stats) + ReLU; per-block scale/shift precompute in LDS,
// then grid-stride elementwise pass. TO_HALF=1: f16 out (feeds GEMM2).
template <int COLS, bool TO_HALF>
__global__ void bn_relu_kernel(const float* __restrict__ in,
                               const float* __restrict__ csum,
                               const float* __restrict__ csq,
                               const float* __restrict__ gamma,
                               const float* __restrict__ beta,
                               _Float16* __restrict__ outh,
                               float* __restrict__ outf,
                               int total) {
    __shared__ float sc[COLS];
    __shared__ float sh[COLS];
    const float invN = 1.0f / (float)N_PTS;
    for (int c = threadIdx.x; c < COLS; c += blockDim.x) {
        float mean  = csum[c] * invN;
        float var   = csq[c] * invN - mean * mean;
        float scale = rsqrtf(var + BN_EPS) * gamma[c];
        sc[c] = scale;
        sh[c] = beta[c] - mean * scale;
    }
    __syncthreads();
    for (int i = blockIdx.x * blockDim.x + threadIdx.x; i < total;
         i += gridDim.x * blockDim.x) {
        const int c = i & (COLS - 1);
        float v = fmaxf(in[i] * sc[c] + sh[c], 0.f);
        if (TO_HALF) outh[i] = (_Float16)v;
        else         outf[i] = v;
    }
}

// ---------------------------------------------------------------------------
extern "C" void kernel_launch(void* const* d_in, const int* in_sizes, int n_in,
                              void* d_out, int out_size, void* d_ws, size_t ws_size,
                              hipStream_t stream) {
    (void)in_sizes; (void)n_in; (void)out_size; (void)ws_size;
    const float* p  = (const float*)d_in[0];
    const float* x  = (const float*)d_in[1];
    const int*  idx = (const int*)d_in[2];
    const float* W1 = (const float*)d_in[3];
    const float* g1 = (const float*)d_in[4];
    const float* b1 = (const float*)d_in[5];
    const float* W2 = (const float*)d_in[6];
    const float* g2 = (const float*)d_in[7];
    const float* b2 = (const float*)d_in[8];
    float* out = (float*)d_out;

    char* w = (char*)d_ws;
    auto carve = [&](size_t bytes) {
        char* q = w;
        w += (bytes + 255) & ~(size_t)255;
        return q;
    };
    _Float16* pooledh = (_Float16*)carve((size_t)N_PTS * KP1 * sizeof(_Float16));
    _Float16* W1h     = (_Float16*)carve((size_t)H1 * KP1 * sizeof(_Float16));
    _Float16* W2h     = (_Float16*)carve((size_t)H2 * H1 * sizeof(_Float16));
    float*    h1      = (float*)   carve((size_t)N_PTS * H1 * sizeof(float));
    _Float16* h1h     = (_Float16*)carve((size_t)N_PTS * H1 * sizeof(_Float16));
    float*    stats   = (float*)   carve(2 * (H1 + H2) * sizeof(float));
    float* sum1 = stats;
    float* sq1  = stats + H1;
    float* sum2 = stats + 2 * H1;
    float* sq2  = stats + 2 * H1 + H2;

    prep_kernel<<<64, 256, 0, stream>>>(W1, W2, W1h, W2h, stats);
    gather_pool_kernel<<<N_PTS / 8, 256, 0, stream>>>(p, x, idx, pooledh);
    gemm1_kernel<<<N_PTS / 128, 256, 0, stream>>>(pooledh, W1h, h1, sum1, sq1);
    bn_relu_kernel<H1, true><<<512, 256, 0, stream>>>(
        h1, sum1, sq1, g1, b1, h1h, nullptr, N_PTS * H1);
    gemm2_kernel<<<N_PTS / 128, 256, 0, stream>>>(h1h, W2h, out, sum2, sq2);
    bn_relu_kernel<H2, false><<<512, 256, 0, stream>>>(
        out, sum2, sq2, g2, b2, nullptr, out, N_PTS * H2);
}
